// Mamba2_27462020890876
// MI455X (gfx1250) — matrix operand order repaired and checked by reading, not compiled, not run on hardware
//
#include <hip/hip_runtime.h>
#include <hip/hip_bf16.h>

// ---------------------------------------------------------------------------
// Mamba2 forward for MI455X (gfx1250, wave32, WMMA).
// Big GEMMs via v_wmma_f32_16x16x32_bf16 (bf16 in, f32 accumulate),
// 256x64 block tile, 8 waves x (32x64) macro-tiles, double-buffered LDS.
// SSD chunked scan / conv / norm in fp32 VALU (≈4% of FLOPs, bandwidth-cheap).
// ---------------------------------------------------------------------------

typedef __bf16 v16bf __attribute__((ext_vector_type(16)));
typedef __bf16 v8bf  __attribute__((ext_vector_type(8)));
typedef float  v8f   __attribute__((ext_vector_type(8)));

#define BATCH   2
#define SEQLEN  4096
#define DMODEL  1024
#define DINNER  2048
#define NHEADS  32
#define HEADDIM 64
#define DSTATE  16
#define CHUNK   64
#define NCHUNK  (SEQLEN / CHUNK)     // 64
#define CONVDIM (DINNER + 2*DSTATE)  // 2080
#define DPROJ   (2*DINNER + 2*DSTATE + NHEADS) // 4160
#define ROWS    (BATCH * SEQLEN)     // 8192

// ---------------- utility ----------------
__device__ __forceinline__ float silu_f(float x) {
    return x / (1.0f + __expf(-x));
}
__device__ __forceinline__ float softplus_f(float x) {
    return (x > 20.0f) ? x : logf(1.0f + __expf(x));
}

// ---------------- f32 -> bf16 conversion ----------------
__global__ void f32_to_bf16_kernel(const float* __restrict__ src,
                                   __bf16* __restrict__ dst, long n) {
    long i = (long)blockIdx.x * blockDim.x + threadIdx.x;
    long stride = (long)gridDim.x * blockDim.x;
    for (; i < n; i += stride) dst[i] = (__bf16)src[i];
}

// ---------------- WMMA bf16 GEMM: C[M,N] = A[M,K] * W[N,K]^T ----------------
// Block tile 256(M) x 64(N), K-step 32, double-buffered LDS.
// 8 waves, each owns a 32(M) x 64(N) macro-tile = 2x4 WMMA 16x16 tiles
// -> 8 v_wmma per wave per K-step behind a single barrier.
// LDS staging feeds A and B in one contiguous k order (k = 16*(lane/16) + s)
// (A row-major 16x32 K-split halves; W N-major/K-contiguous).
#define GBM 256
#define GBN 64
#define GBK 32
#define GLD (GBK + 8)   // 80B row stride, bank-skewed

__global__ __launch_bounds__(256)
void gemm_bf16_wmma_kernel(const __bf16* __restrict__ A,
                           const __bf16* __restrict__ W,
                           float* __restrict__ C,
                           int M, int N, int K) {
    __shared__ __align__(16) __bf16 As[2][GBM][GLD];  // 2 x 20 KB
    __shared__ __align__(16) __bf16 Ws[2][GBN][GLD];  // 2 x  5 KB

    const int t    = threadIdx.x;
    const int lane = t & 31;
    const int wave = t >> 5;     // 0..7 : 32-row strip of the 256-row tile

    const long blockM = (long)blockIdx.x * GBM;
    const long blockN = (long)blockIdx.y * GBN;

    v8f acc[2][4];
    for (int i = 0; i < 2; ++i)
        for (int j = 0; j < 4; ++j)
            for (int r = 0; r < 8; ++r) acc[i][j][r] = 0.0f;

    // global->LDS mapping: A tile 256x32 (one full row / thread, 4 x v8bf),
    //                      W tile 64x32  (quarter row / thread, 1 x v8bf)
    const int wrow = t >> 2;             // 0..63
    const int wcol = (t & 3) * 8;        // 0,8,16,24
    const __bf16* aptr = A + (blockM + t) * (long)K;
    const __bf16* wptr = W + (blockN + wrow) * (long)K + wcol;

    const int rA    = lane & 15;
    const int halfA = (lane < 16) ? 0 : 16;  // contiguous K-halves (k = 16h + s), same as B
    const int halfB = (lane < 16) ? 0 : 16;  // contiguous K-halves, same order as A
    const int mbase = wave * 32;

    // ---- preload tile 0 into buffer 0 ----
    {
        v8bf a0 = *(const v8bf*)(aptr);
        v8bf a1 = *(const v8bf*)(aptr + 8);
        v8bf a2 = *(const v8bf*)(aptr + 16);
        v8bf a3 = *(const v8bf*)(aptr + 24);
        v8bf w0 = *(const v8bf*)(wptr);
        *(v8bf*)&As[0][t][0]        = a0;
        *(v8bf*)&As[0][t][8]        = a1;
        *(v8bf*)&As[0][t][16]       = a2;
        *(v8bf*)&As[0][t][24]       = a3;
        *(v8bf*)&Ws[0][wrow][wcol]  = w0;
    }

    int cur = 0;
    for (int k0 = 0; k0 < K; k0 += GBK) {
        __syncthreads();   // publishes buffer `cur`, retires reads of `cur^1`
        const bool hasNext = (k0 + GBK) < K;

        // stage next K tile in registers while computing the current one
        v8bf na0, na1, na2, na3, nw0;
        if (hasNext) {
            const __bf16* ap = aptr + k0 + GBK;
            const __bf16* wp = wptr + k0 + GBK;
            na0 = *(const v8bf*)(ap);
            na1 = *(const v8bf*)(ap + 8);
            na2 = *(const v8bf*)(ap + 16);
            na3 = *(const v8bf*)(ap + 24);
            nw0 = *(const v8bf*)(wp);
            if (k0 + 2 * GBK < K) {  // keep L2 warm two tiles ahead
                __builtin_prefetch(ap + GBK, 0, 1);
                __builtin_prefetch(wp + GBK, 0, 1);
            }
        }

        // ---- fragments + 8 WMMA from buffer `cur` ----
        v16bf afrag[2], bfrag[4];
        for (int mt = 0; mt < 2; ++mt) {
            const __bf16* p = &As[cur][mbase + mt * 16 + rA][halfA];
            v8bf lo = *(const v8bf*)p;          // K 0..7  (or 16..23)
            v8bf hi = *(const v8bf*)(p + 8);    // K 8..15 (or 24..31)
            afrag[mt] = __builtin_shufflevector(lo, hi,
                0,1,2,3,4,5,6,7,8,9,10,11,12,13,14,15);
        }
        for (int nt = 0; nt < 4; ++nt) {
            const __bf16* p = &Ws[cur][nt * 16 + rA][halfB];
            v8bf lo = *(const v8bf*)p;          // 16 contiguous K of column N
            v8bf hi = *(const v8bf*)(p + 8);
            bfrag[nt] = __builtin_shufflevector(lo, hi,
                0,1,2,3,4,5,6,7,8,9,10,11,12,13,14,15);
        }
        for (int mt = 0; mt < 2; ++mt)
            for (int nt = 0; nt < 4; ++nt)
                acc[mt][nt] = __builtin_amdgcn_wmma_f32_16x16x32_bf16(
                    /*neg_a=*/false, afrag[mt],
                    /*neg_b=*/false, bfrag[nt],
                    /*c_mod=*/(short)0, acc[mt][nt],
                    /*reuse_a=*/false, /*reuse_b=*/false);

        // ---- stash next tile into the other buffer ----
        if (hasNext) {
            const int nxt = cur ^ 1;
            *(v8bf*)&As[nxt][t][0]       = na0;
            *(v8bf*)&As[nxt][t][8]       = na1;
            *(v8bf*)&As[nxt][t][16]      = na2;
            *(v8bf*)&As[nxt][t][24]      = na3;
            *(v8bf*)&Ws[nxt][wrow][wcol] = nw0;
        }
        cur ^= 1;
    }

    // D layout: lane%16 = N, VGPR r = M row r (+8 for lanes 16..31)
    const int rowOff  = (lane >= 16) ? 8 : 0;
    const int colLane = lane & 15;
    for (int mt = 0; mt < 2; ++mt)
        for (int nt = 0; nt < 4; ++nt)
            for (int r = 0; r < 8; ++r) {
                long grow = blockM + mbase + mt * 16 + r + rowOff;
                long gcol = blockN + nt * 16 + colLane;
                C[grow * N + gcol] = acc[mt][nt][r];
            }
}

// ---------------- causal depthwise conv(4) + bias + SiLU ----------------
__global__ void conv_silu_kernel(const float* __restrict__ zxbcdt,
                                 const float* __restrict__ conv_w,
                                 const float* __restrict__ conv_b,
                                 float* __restrict__ xbc) {
    long idx = (long)blockIdx.x * blockDim.x + threadIdx.x;
    const long total = (long)ROWS * CONVDIM;
    if (idx >= total) return;
    int ch  = (int)(idx % CONVDIM);
    long bl = idx / CONVDIM;
    int l   = (int)(bl % SEQLEN);
    long brow = bl - l;  // batch row base
    float acc = conv_b[ch];
    #pragma unroll
    for (int k = 0; k < 4; ++k) {
        int ls = l - 3 + k;
        if (ls >= 0)
            acc += conv_w[ch * 4 + k] *
                   zxbcdt[(brow + ls) * (long)DPROJ + DINNER + ch];
    }
    xbc[idx] = silu_f(acc);
}

// ---------------- dt = softplus(raw + bias) ----------------
__global__ void dt_kernel(const float* __restrict__ zxbcdt,
                          const float* __restrict__ dt_bias,
                          float* __restrict__ dt) {
    long idx = (long)blockIdx.x * blockDim.x + threadIdx.x;
    const long total = (long)ROWS * NHEADS;
    if (idx >= total) return;
    int h = (int)(idx % NHEADS);
    long bl = idx / NHEADS;
    dt[idx] = softplus_f(zxbcdt[bl * (long)DPROJ + (DPROJ - NHEADS) + h] + dt_bias[h]);
}

// ---------------- SSD pass 1: per (b, chunk, head) ----------------
// cumsum(a), L-matrix, Y_diag, per-chunk states.
__global__ __launch_bounds__(256)
void ssd1_kernel(const float* __restrict__ xbc,
                 const float* __restrict__ dt,
                 const float* __restrict__ A_log,
                 float* __restrict__ ybuf,
                 float* __restrict__ cumbuf,
                 float* __restrict__ cstates) {
    const int blk = blockIdx.x;                 // ((b*NCHUNK)+c)*NHEADS + h
    const int h = blk % NHEADS;
    const int c = (blk / NHEADS) % NCHUNK;
    const int b = blk / (NHEADS * NCHUNK);
    const int t = threadIdx.x;
    const long rowbase = (long)b * SEQLEN + (long)c * CHUNK;

    __shared__ float xs[CHUNK][HEADDIM + 1];    // x * dt
    __shared__ float Bs[CHUNK][DSTATE];
    __shared__ float Cs[CHUNK][DSTATE];
    __shared__ float G[CHUNK][CHUNK + 1];
    __shared__ float avals[CHUNK];
    __shared__ float cum[CHUNK];
    __shared__ float decay[CHUNK];

    const float Ah = -__expf(A_log[h]);

    for (int i = t; i < CHUNK * HEADDIM; i += 256) {
        int l = i >> 6, p = i & 63;
        float d = dt[(rowbase + l) * NHEADS + h];
        xs[l][p] = xbc[(rowbase + l) * (long)CONVDIM + h * HEADDIM + p] * d;
    }
    for (int i = t; i < CHUNK * DSTATE; i += 256) {
        int l = i >> 4, n = i & 15;
        Bs[l][n] = xbc[(rowbase + l) * (long)CONVDIM + DINNER + n];
        Cs[l][n] = xbc[(rowbase + l) * (long)CONVDIM + DINNER + DSTATE + n];
    }
    if (t < CHUNK) avals[t] = Ah * dt[(rowbase + t) * NHEADS + h];
    __syncthreads();
    if (t == 0) {
        float s = 0.0f;
        for (int l = 0; l < CHUNK; ++l) { s += avals[l]; cum[l] = s; }
    }
    __syncthreads();
    if (t < CHUNK) decay[t] = __expf(cum[CHUNK - 1] - cum[t]);
    __syncthreads();

    // G[l][s] = (C_l . B_s) * exp(cum_l - cum_s)  for s <= l
    for (int i = t; i < CHUNK * CHUNK; i += 256) {
        int l = i >> 6, s = i & 63;
        float g = 0.0f;
        if (s <= l) {
            float d = 0.0f;
            #pragma unroll
            for (int n = 0; n < DSTATE; ++n) d += Cs[l][n] * Bs[s][n];
            g = d * __expf(cum[l] - cum[s]);
        }
        G[l][s] = g;
    }
    __syncthreads();

    // Y_diag
    for (int i = t; i < CHUNK * HEADDIM; i += 256) {
        int l = i >> 6, p = i & 63;
        float acc = 0.0f;
        for (int s = 0; s <= l; ++s) acc += G[l][s] * xs[s][p];
        ybuf[(rowbase + l) * (long)DINNER + h * HEADDIM + p] = acc;
    }
    // chunk states[p][n]
    for (int i = t; i < HEADDIM * DSTATE; i += 256) {
        int p = i >> 4, n = i & 15;
        float acc = 0.0f;
        for (int l = 0; l < CHUNK; ++l) acc += Bs[l][n] * decay[l] * xs[l][p];
        cstates[(long)blk * (HEADDIM * DSTATE) + i] = acc;
    }
    if (t < CHUNK) cumbuf[(long)blk * CHUNK + t] = cum[t];
}

// ---------------- SSD pass 2: sequential inter-chunk recurrence ----------------
__global__ __launch_bounds__(256)
void ssd2_kernel(const float* __restrict__ cstates,
                 const float* __restrict__ cumbuf,
                 float* __restrict__ sprefix) {
    const int blk = blockIdx.x;        // b*NHEADS + h
    const int h = blk % NHEADS;
    const int b = blk / NHEADS;
    const int t = threadIdx.x;         // 256 threads, 4 state elems each
    float s[4] = {0.f, 0.f, 0.f, 0.f};
    for (int c = 0; c < NCHUNK; ++c) {
        long g = (((long)b * NCHUNK + c) * NHEADS + h);
        long base = g * (HEADDIM * DSTATE);
        #pragma unroll
        for (int j = 0; j < 4; ++j) sprefix[base + t + j * 256] = s[j];
        float dec = __expf(cumbuf[g * CHUNK + (CHUNK - 1)]);
        #pragma unroll
        for (int j = 0; j < 4; ++j)
            s[j] = s[j] * dec + cstates[base + t + j * 256];
    }
}

// ---------------- SSD pass 3: Y_off + D skip ----------------
__global__ __launch_bounds__(256)
void ssd3_kernel(const float* __restrict__ xbc,
                 const float* __restrict__ cumbuf,
                 const float* __restrict__ sprefix,
                 const float* __restrict__ Dv,
                 float* __restrict__ ybuf) {
    const int blk = blockIdx.x;
    const int h = blk % NHEADS;
    const int c = (blk / NHEADS) % NCHUNK;
    const int b = blk / (NHEADS * NCHUNK);
    const int t = threadIdx.x;
    const long rowbase = (long)b * SEQLEN + (long)c * CHUNK;

    __shared__ float S[HEADDIM][DSTATE];
    __shared__ float Cs[CHUNK][DSTATE];
    __shared__ float sdo[CHUNK];

    for (int i = t; i < HEADDIM * DSTATE; i += 256)
        S[i >> 4][i & 15] = sprefix[(long)blk * (HEADDIM * DSTATE) + i];
    for (int i = t; i < CHUNK * DSTATE; i += 256) {
        int l = i >> 4, n = i & 15;
        Cs[l][n] = xbc[(rowbase + l) * (long)CONVDIM + DINNER + DSTATE + n];
    }
    if (t < CHUNK) sdo[t] = __expf(cumbuf[(long)blk * CHUNK + t]);
    __syncthreads();

    const float Dh = Dv[h];
    for (int i = t; i < CHUNK * HEADDIM; i += 256) {
        int l = i >> 6, p = i & 63;
        float acc = 0.0f;
        #pragma unroll
        for (int n = 0; n < DSTATE; ++n) acc += Cs[l][n] * S[p][n];
        long row = rowbase + l;
        float xval = xbc[row * (long)CONVDIM + h * HEADDIM + p];
        ybuf[row * (long)DINNER + h * HEADDIM + p] += sdo[l] * acc + Dh * xval;
    }
}

// ---------------- gate (silu(z)) + RMSNorm -> bf16 ----------------
__global__ __launch_bounds__(256)
void gate_norm_kernel(const float* __restrict__ ybuf,
                      const float* __restrict__ zxbcdt,
                      const float* __restrict__ norm_w,
                      __bf16* __restrict__ ybf) {
    const long row = blockIdx.x;      // 0..ROWS-1
    const int t = threadIdx.x;
    __shared__ float red[256];
    float vals[8];
    float ss = 0.0f;
    #pragma unroll
    for (int j = 0; j < 8; ++j) {
        int col = t + j * 256;
        float z = zxbcdt[row * (long)DPROJ + col];   // z block = first DINNER cols
        float v = ybuf[row * (long)DINNER + col] * silu_f(z);
        vals[j] = v;
        ss += v * v;
    }
    red[t] = ss;
    __syncthreads();
    for (int off = 128; off > 0; off >>= 1) {
        if (t < off) red[t] += red[t + off];
        __syncthreads();
    }
    const float r = rsqrtf(red[0] / (float)DINNER + 1e-5f);
    #pragma unroll
    for (int j = 0; j < 8; ++j) {
        int col = t + j * 256;
        ybf[row * (long)DINNER + col] = (__bf16)(vals[j] * r * norm_w[col]);
    }
}

// ---------------------------------------------------------------------------
// Host launcher
// ---------------------------------------------------------------------------
extern "C" void kernel_launch(void* const* d_in, const int* in_sizes, int n_in,
                              void* d_out, int out_size, void* d_ws, size_t ws_size,
                              hipStream_t stream) {
    const float* u       = (const float*)d_in[0];  // (B,L,DMODEL)
    const float* W_in    = (const float*)d_in[1];  // (DPROJ, DMODEL)
    const float* conv_w  = (const float*)d_in[2];  // (CONVDIM,1,4)
    const float* conv_b  = (const float*)d_in[3];  // (CONVDIM,)
    const float* dt_bias = (const float*)d_in[4];  // (NHEADS,)
    const float* A_log   = (const float*)d_in[5];  // (NHEADS,)
    const float* Dv      = (const float*)d_in[6];  // (NHEADS,)
    const float* norm_w  = (const float*)d_in[7];  // (DINNER,)
    const float* W_out   = (const float*)d_in[8];  // (DMODEL, DINNER)
    float* out = (float*)d_out;                    // (B,L,DMODEL)

    char* ws = (char*)d_ws;
    size_t off = 0;
    auto carve = [&](size_t bytes) { char* p = ws + off; off += (bytes + 255) & ~size_t(255); return p; };
    __bf16* ubf      = (__bf16*)carve((size_t)ROWS * DMODEL * 2);
    __bf16* wibf     = (__bf16*)carve((size_t)DPROJ * DMODEL * 2);
    float*  zxbcdt   = (float*) carve((size_t)ROWS * DPROJ * 4);
    float*  xbc      = (float*) carve((size_t)ROWS * CONVDIM * 4);
    float*  dt       = (float*) carve((size_t)ROWS * NHEADS * 4);
    float*  cumbuf   = (float*) carve((size_t)BATCH * NCHUNK * NHEADS * CHUNK * 4);
    float*  cstates  = (float*) carve((size_t)BATCH * NCHUNK * NHEADS * HEADDIM * DSTATE * 4);
    float*  sprefix  = (float*) carve((size_t)BATCH * NCHUNK * NHEADS * HEADDIM * DSTATE * 4);
    float*  ybuf     = (float*) carve((size_t)ROWS * DINNER * 4);
    __bf16* ybf      = (__bf16*)carve((size_t)ROWS * DINNER * 2);
    __bf16* wobf     = (__bf16*)carve((size_t)DMODEL * DINNER * 2);
    (void)ws_size; (void)n_in; (void)in_sizes; (void)out_size;

    // 1) bf16 conversions for GEMM operands
    f32_to_bf16_kernel<<<2048, 256, 0, stream>>>(u,     ubf,  (long)ROWS * DMODEL);
    f32_to_bf16_kernel<<<2048, 256, 0, stream>>>(W_in,  wibf, (long)DPROJ * DMODEL);
    f32_to_bf16_kernel<<<2048, 256, 0, stream>>>(W_out, wobf, (long)DMODEL * DINNER);

    // 2) in_proj GEMM: zxbcdt[ROWS, DPROJ] = u * W_in^T
    {
        dim3 grid(ROWS / GBM, DPROJ / GBN);   // 32 x 65
        gemm_bf16_wmma_kernel<<<grid, 256, 0, stream>>>(ubf, wibf, zxbcdt,
                                                        ROWS, DPROJ, DMODEL);
    }

    // 3) causal conv + SiLU over xBC slice
    {
        long total = (long)ROWS * CONVDIM;
        conv_silu_kernel<<<(int)((total + 255) / 256), 256, 0, stream>>>(
            zxbcdt, conv_w, conv_b, xbc);
    }

    // 4) dt = softplus(. + bias)
    {
        long total = (long)ROWS * NHEADS;
        dt_kernel<<<(int)((total + 255) / 256), 256, 0, stream>>>(zxbcdt, dt_bias, dt);
    }

    // 5-7) SSD chunked scan
    const int nblk = BATCH * NCHUNK * NHEADS;  // 4096
    ssd1_kernel<<<nblk, 256, 0, stream>>>(xbc, dt, A_log, ybuf, cumbuf, cstates);
    ssd2_kernel<<<BATCH * NHEADS, 256, 0, stream>>>(cstates, cumbuf, sprefix);
    ssd3_kernel<<<nblk, 256, 0, stream>>>(xbc, cumbuf, sprefix, Dv, ybuf);

    // 8) gate + RMSNorm, emit bf16 activations for out_proj
    gate_norm_kernel<<<ROWS, 256, 0, stream>>>(ybuf, zxbcdt, norm_w, ybf);

    // 9) out_proj GEMM: out[ROWS, DMODEL] = y * W_out^T
    {
        dim3 grid(ROWS / GBM, DMODEL / GBN);  // 32 x 16
        gemm_bf16_wmma_kernel<<<grid, 256, 0, stream>>>(ybf, wobf, out,
                                                        ROWS, DMODEL, DINNER);
    }
}
